// InternLM2Attention_13365938225663
// MI455X (gfx1250) — compile-verified, matrix operand order checked by reading, not run on hardware
//
#include <hip/hip_runtime.h>
#include <hip/hip_bf16.h>

// ---------------------------------------------------------------------------
// InternLM2 attention block for gfx1250 (MI455X), bf16 WMMA pipeline.
//   1) cast hidden -> bf16 ; transpose-cast w_qkv, w_o -> bf16 N-major
//   2) QKV GEMM: TDM ping-pong staging (tensor_load_to_lds) + WMMA compute
//   3) RoPE + split into q/k (row-major) and vT (d-major)
//   4) GQA causal flash attention; K/V tiles double-buffered via TDM
//   5) O GEMM -> fp32 d_out
// ---------------------------------------------------------------------------

typedef __bf16 bf16;
typedef __attribute__((ext_vector_type(16))) __bf16 v16bf;
typedef __attribute__((ext_vector_type(8)))  float  v8f;

#define S_LEN 2048
#define HID   4096
#define NQ_H  32
#define NKV_H 8
#define HD    128
#define FQKV  6144   // (NQ + 2*NKV) * D
#define QDIM  4096   // NQ * D
#define KVDIM 1024   // NKV * D
#define GQA_GROUP 4  // NQ / NKV

// ----------------------------- TDM support ----------------------------------

#if __has_builtin(__builtin_amdgcn_tensor_load_to_lds) && \
    __has_builtin(__builtin_amdgcn_s_wait_tensorcnt)
#define HAVE_TDM 1
#else
#define HAVE_TDM 0
#endif

#if HAVE_TDM
typedef unsigned int u32x4 __attribute__((ext_vector_type(4)));
typedef int          i32x4 __attribute__((ext_vector_type(4)));
typedef int          i32x8 __attribute__((ext_vector_type(8)));

__device__ inline unsigned lds_off(const void* p) {
  // addrspacecast generic->LDS(3) yields the 32-bit LDS offset for D#.lds_addr
  return (unsigned)(unsigned long long)(__attribute__((address_space(3))) const void*)p;
}

__device__ inline void tdm_load(u32x4 g0, i32x8 g1) {
  i32x4 z4 = {0, 0, 0, 0};
#if __has_include(<hip/amd_detail/amd_gfx1250_TDM.h>)
  i32x8 z8 = {0, 0, 0, 0, 0, 0, 0, 0};
  __builtin_amdgcn_tensor_load_to_lds(g0, g1, z4, z4, z8, 0);  // 6-arg toolchain
#else
  __builtin_amdgcn_tensor_load_to_lds(g0, g1, z4, z4, 0);      // 5-arg ROCm
#endif
}

// D# group0: count=1, lds byte addr, 57-bit global byte addr, type=2 ("image")
__device__ inline u32x4 tdm_g0(unsigned lds_addr, unsigned long long ga) {
  u32x4 g0;
  g0[0] = 1u;
  g0[1] = lds_addr;
  g0[2] = (unsigned)ga;
  g0[3] = (unsigned)(ga >> 32) | (2u << 30);
  return g0;
}

// D# group1 for a 2D bf16 tile with LDS padding:
//   w0: data_size=2B (1<<16) | pad_enable (1<<20) | pad_interval<<22 | pad_amount<<25
//   tensor_dim0[63:48]->w1 hi16, tensor_dim1[111:80]->w2 hi16 (dims < 64K here)
//   tile_dim0[127:112]->w3 hi16, tile_dim1[143:128]->w4 lo16
//   tensor_dim0_stride[207:160]->w5 (lo32)
__device__ inline i32x8 tdm_g1_2d(int tensor_d0, int tensor_d1, int tile_d0,
                                  int tile_d1, int stride0, int pad_interval_code,
                                  int pad_amount_code) {
  i32x8 g1;
  g1[0] = (1 << 16) | (1 << 20) | (pad_interval_code << 22) | (pad_amount_code << 25);
  g1[1] = tensor_d0 << 16;
  g1[2] = tensor_d1 << 16;
  g1[3] = tile_d0 << 16;
  g1[4] = tile_d1 & 0xFFFF;
  g1[5] = stride0;
  g1[6] = 0;
  g1[7] = 0;
  return g1;
}
#endif  // HAVE_TDM

// ---------------- WMMA helpers (layouts per CDNA5 ISA 7.12.2) ---------------

__device__ inline v8f wmma_bf16(v16bf a, v16bf b, v8f c) {
  return __builtin_amdgcn_wmma_f32_16x16x32_bf16(
      false, a, false, b, (short)0, c, false, false);
}

__device__ inline v8f zero_v8f() {
  v8f z;
#pragma unroll
  for (int i = 0; i < 8; ++i) z[i] = 0.0f;
  return z;
}

// A fragment: 16x32 (MxK), K contiguous per row; two 16B runs -> ds_load_b128.
__device__ inline v16bf load_a_frag(const bf16* p, int ld) {
  const int lane = threadIdx.x & 31;
  const int r    = lane & 15;
  const int kb   = (lane >> 4) * 8;
  v16bf f;
#pragma unroll
  for (int i = 0; i < 8; ++i) {
    f[i]     = p[r * ld + kb + i];
    f[i + 8] = p[r * ld + kb + 16 + i];
  }
  return f;
}

// B fragment from an N-major buffer: source is [N][K], K contiguous per column.
__device__ inline v16bf load_bt_frag(const bf16* p, int ld) {
  const int lane = threadIdx.x & 31;
  const int c    = lane & 15;
  const int kb   = (lane >> 4) * 8;
  v16bf f;
#pragma unroll
  for (int i = 0; i < 8; ++i) {
    f[i]     = p[c * ld + kb + i];
    f[i + 8] = p[c * ld + kb + 16 + i];
  }
  return f;
}

// ---------------------------- casts / transpose -----------------------------

__global__ __launch_bounds__(256) void cast_f32_bf16(const float* __restrict__ src,
                                                     bf16* __restrict__ dst, int n) {
  for (int i = blockIdx.x * blockDim.x + threadIdx.x; i < n;
       i += gridDim.x * blockDim.x) {
    dst[i] = (bf16)src[i];
  }
}

// dst[C][R] (bf16) = transpose(src[R][C] (f32)); coalesced via LDS tile.
__global__ __launch_bounds__(256) void transpose_cast(const float* __restrict__ src,
                                                      bf16* __restrict__ dst,
                                                      int R, int C) {
  __shared__ float tile[32][33];
  const int tx = threadIdx.x & 31, ty = threadIdx.x >> 5;
  const int r0 = blockIdx.y * 32, c0 = blockIdx.x * 32;
#pragma unroll
  for (int j = 0; j < 32; j += 8)
    tile[ty + j][tx] = src[(size_t)(r0 + ty + j) * C + c0 + tx];
  __syncthreads();
#pragma unroll
  for (int j = 0; j < 32; j += 8)
    dst[(size_t)(c0 + ty + j) * R + r0 + tx] = (bf16)tile[tx][ty + j];
}

// ------------------------------ bf16 GEMM (TN) -------------------------------
// C[M,N] (f32) = A[M,K] (bf16 row-major) * Bt[N,K] (bf16 N-major)
// Block: 256 threads = 8 waves; block tile 64x256; wave tile 32x64 (2x4 WMMA).
// TDM path: ping-pong LDS buffers; wave 0 issues the next K-tile's DMA right
// after the barrier, so the Tensor Data Mover runs under the WMMA work and
// only ONE barrier per K-tile is needed.

#define LDT 40  // 32 + 8 bf16 pad (TDM: pad_interval=16DW code 3, pad=4DW code 3)

__global__ __launch_bounds__(256) void gemm_bf16_tn(const bf16* __restrict__ A,
                                                    const bf16* __restrict__ Bt,
                                                    float* __restrict__ C,
                                                    int M, int N, int K) {
#if HAVE_TDM
  __shared__ __attribute__((aligned(16))) bf16 As[2][64 * LDT];
  __shared__ __attribute__((aligned(16))) bf16 Bs[2][256 * LDT];
#else
  __shared__ __attribute__((aligned(16))) bf16 As[1][64 * LDT];
  __shared__ __attribute__((aligned(16))) bf16 Bs[1][256 * LDT];
#endif

  const int tid  = threadIdx.x;
  const int wave = tid >> 5;
  const int lane = tid & 31;
  const int row0 = blockIdx.y * 64;
  const int col0 = blockIdx.x * 256;
  const int wr   = wave >> 2;  // 0..1 -> 32 rows
  const int wc   = wave & 3;   // 0..3 -> 64 cols

  v8f acc[2][4];
#pragma unroll
  for (int i = 0; i < 2; ++i)
#pragma unroll
    for (int j = 0; j < 4; ++j) acc[i][j] = zero_v8f();

#if HAVE_TDM
  const unsigned lds_a[2] = {lds_off(&As[0][0]), lds_off(&As[1][0])};
  const unsigned lds_b[2] = {lds_off(&Bs[0][0]), lds_off(&Bs[1][0])};
  if (wave == 0) {  // prologue: stage K-tile 0 into set 0
    unsigned long long gaA = (unsigned long long)(const void*)&A[(size_t)row0 * K];
    unsigned long long gaB = (unsigned long long)(const void*)&Bt[(size_t)col0 * K];
    tdm_load(tdm_g0(lds_a[0], gaA), tdm_g1_2d(K, M, 32, 64, K, 3, 3));
    tdm_load(tdm_g0(lds_b[0], gaB), tdm_g1_2d(K, N, 32, 256, K, 3, 3));
  }
#else
  const int ar = tid >> 2, ac = (tid & 3) * 8;
#endif

  for (int k0 = 0; k0 < K; k0 += 32) {
#if HAVE_TDM
    const int buf = (k0 >> 5) & 1;
    if (wave == 0) __builtin_amdgcn_s_wait_tensorcnt(0);
    __syncthreads();  // tile `buf` resident; alt buffer's readers all done
    if (wave == 0 && k0 + 32 < K) {  // stage next tile into alt set (DMA
      const int nxt = buf ^ 1;       //  overlaps this tile's WMMA work)
      unsigned long long gaA =
          (unsigned long long)(const void*)&A[(size_t)row0 * K + k0 + 32];
      unsigned long long gaB =
          (unsigned long long)(const void*)&Bt[(size_t)col0 * K + k0 + 32];
      tdm_load(tdm_g0(lds_a[nxt], gaA), tdm_g1_2d(K, M, 32, 64, K, 3, 3));
      tdm_load(tdm_g0(lds_b[nxt], gaB), tdm_g1_2d(K, N, 32, 256, K, 3, 3));
    }
    const bf16* Ap = &As[buf][0];
    const bf16* Bp = &Bs[buf][0];
#else
    uint4 ta = *(const uint4*)&A[(row0 + ar) * K + k0 + ac];
    uint4 tb[4];
#pragma unroll
    for (int j = 0; j < 4; ++j)
      tb[j] = *(const uint4*)&Bt[(size_t)(col0 + tid) * K + k0 + j * 8];
    if (k0 + 32 < K) {
      __builtin_prefetch(&A[(row0 + ar) * K + k0 + 32 + ac], 0, 1);
      __builtin_prefetch(&Bt[(size_t)(col0 + tid) * K + k0 + 32], 0, 1);
    }
    __syncthreads();
    *(uint4*)&As[0][ar * LDT + ac] = ta;
#pragma unroll
    for (int j = 0; j < 4; ++j) *(uint4*)&Bs[0][tid * LDT + j * 8] = tb[j];
    __syncthreads();
    const bf16* Ap = &As[0][0];
    const bf16* Bp = &Bs[0][0];
#endif

    v16bf a0 = load_a_frag(&Ap[(wr * 32) * LDT], LDT);
    v16bf a1 = load_a_frag(&Ap[(wr * 32 + 16) * LDT], LDT);
#pragma unroll
    for (int ni = 0; ni < 4; ++ni) {
      v16bf b = load_bt_frag(&Bp[(wc * 64 + ni * 16) * LDT], LDT);
      acc[0][ni] = wmma_bf16(a0, b, acc[0][ni]);
      acc[1][ni] = wmma_bf16(a1, b, acc[1][ni]);
    }
  }

  const int n = lane & 15, g = lane >> 4;
#pragma unroll
  for (int mi = 0; mi < 2; ++mi)
#pragma unroll
    for (int ni = 0; ni < 4; ++ni)
#pragma unroll
      for (int i = 0; i < 8; ++i) {
        C[(size_t)(row0 + wr * 32 + mi * 16 + i + 8 * g) * N +
          col0 + wc * 64 + ni * 16 + n] = acc[mi][ni][i];
      }
}

// --------------------------- RoPE + QKV split -------------------------------

__global__ __launch_bounds__(256) void rope_split(const float* __restrict__ qkv,
                                                  const float* __restrict__ cos_t,
                                                  const float* __restrict__ sin_t,
                                                  bf16* __restrict__ q,
                                                  bf16* __restrict__ k,
                                                  bf16* __restrict__ vT) {
  const int total = S_LEN * FQKV;
  for (int idx = blockIdx.x * blockDim.x + threadIdx.x; idx < total;
       idx += gridDim.x * blockDim.x) {
    const int s = idx / FQKV;
    const int f = idx - s * FQKV;
    const float x = qkv[idx];
    if (f < QDIM) {  // Q with RoPE
      const int d = f & (HD - 1);
      const float c  = cos_t[s * HD + d];
      const float sn = sin_t[s * HD + d];
      const float part = (d < HD / 2) ? -qkv[idx + HD / 2] : qkv[idx - HD / 2];
      q[s * QDIM + f] = (bf16)(x * c + part * sn);
    } else if (f < QDIM + KVDIM) {  // K with RoPE
      const int fl = f - QDIM;
      const int d  = fl & (HD - 1);
      const float c  = cos_t[s * HD + d];
      const float sn = sin_t[s * HD + d];
      const float part = (d < HD / 2) ? -qkv[idx + HD / 2] : qkv[idx - HD / 2];
      k[s * KVDIM + fl] = (bf16)(x * c + part * sn);
    } else {  // V plain, transposed store (d-major)
      vT[(f - QDIM - KVDIM) * S_LEN + s] = (bf16)x;
    }
  }
}

// --------------------------- flash attention --------------------------------
// grid = (S/128, NQ). Block: 8 waves; wave w owns query rows [blk*128+w*16, +16).
// Keys processed in chunks of 32 (causal). K chunk [key][d], V chunk [d][key].
// TDM path: double-buffered; the next chunk's DMA runs under this chunk's
// softmax + WMMA work.

#define LDKV 136  // 128 + 8 pad (row = 64 DW -> interval code 5, pad 4 DW code 3)
#define LDVT 40   // 32 + 8 pad  (row = 16 DW -> interval code 3, pad 4 DW code 3)
#define LDP  36   // 32 + 4 pad  (P scratch)

__global__ __launch_bounds__(256) void flash_attn(const bf16* __restrict__ q,
                                                  const bf16* __restrict__ k,
                                                  const bf16* __restrict__ vT,
                                                  bf16* __restrict__ attn) {
#if HAVE_TDM
  __shared__ __attribute__((aligned(16))) bf16 Ks[2][32 * LDKV];
  __shared__ __attribute__((aligned(16))) bf16 Vs[2][128 * LDVT];
#else
  __shared__ __attribute__((aligned(16))) bf16 Ks[1][32 * LDKV];
  __shared__ __attribute__((aligned(16))) bf16 Vs[1][128 * LDVT];
#endif
  __shared__ __attribute__((aligned(16))) bf16 Ps[8 * 16 * LDP];

  const int h    = blockIdx.y;
  const int kvh  = h / GQA_GROUP;
  const int qblk = blockIdx.x;
  const int tid  = threadIdx.x;
  const int wave = tid >> 5;
  const int lane = tid & 31;
  const int qrow0 = qblk * 128 + wave * 16;
  bf16* myP = &Ps[wave * 16 * LDP];

  v16bf qf[4];
#pragma unroll
  for (int d = 0; d < 4; ++d)
    qf[d] = load_a_frag(q + qrow0 * QDIM + h * HD + d * 32, QDIM);

  v8f o[8];
  float m_i[8], l_i[8];
#pragma unroll
  for (int dc = 0; dc < 8; ++dc) o[dc] = zero_v8f();
#pragma unroll
  for (int i = 0; i < 8; ++i) { m_i[i] = -3.0e38f; l_i[i] = 0.0f; }

  const int n = lane & 15, g = lane >> 4;
  const float scale = 0.08838834764831845f;  // 1/sqrt(128)
  const int nch = qblk * 4 + 4;

#if HAVE_TDM
  const unsigned lds_k[2] = {lds_off(&Ks[0][0]), lds_off(&Ks[1][0])};
  const unsigned lds_v[2] = {lds_off(&Vs[0][0]), lds_off(&Vs[1][0])};
  if (wave == 0) {  // prologue: chunk 0 into set 0
    unsigned long long gaK = (unsigned long long)(const void*)&k[kvh * HD];
    unsigned long long gaV =
        (unsigned long long)(const void*)&vT[(size_t)(kvh * HD) * S_LEN];
    tdm_load(tdm_g0(lds_k[0], gaK), tdm_g1_2d(KVDIM, S_LEN, HD, 32, KVDIM, 5, 3));
    tdm_load(tdm_g0(lds_v[0], gaV), tdm_g1_2d(S_LEN, KVDIM, 32, HD, S_LEN, 3, 3));
  }
#else
  const int kr = tid >> 3, kc = (tid & 7) * 16;
  const int vd = tid >> 1, vh = (tid & 1) * 16;
#endif

  for (int ch = 0; ch < nch; ++ch) {
    const int kb = ch * 32;

#if HAVE_TDM
    const int buf = ch & 1;
    if (wave == 0) __builtin_amdgcn_s_wait_tensorcnt(0);
    __syncthreads();  // chunk `buf` resident; alt set's readers all done
    if (wave == 0 && ch + 1 < nch) {  // stage next chunk into alt set
      const int nxt = buf ^ 1;
      unsigned long long gaK =
          (unsigned long long)(const void*)&k[(size_t)(kb + 32) * KVDIM + kvh * HD];
      unsigned long long gaV =
          (unsigned long long)(const void*)&vT[(size_t)(kvh * HD) * S_LEN + kb + 32];
      tdm_load(tdm_g0(lds_k[nxt], gaK),
               tdm_g1_2d(KVDIM, S_LEN, HD, 32, KVDIM, 5, 3));
      tdm_load(tdm_g0(lds_v[nxt], gaV),
               tdm_g1_2d(S_LEN, KVDIM, 32, HD, S_LEN, 3, 3));
    }
    const bf16* Kc = &Ks[buf][0];
    const bf16* Vc = &Vs[buf][0];
#else
    uint4 tk0 = *(const uint4*)&k[(kb + kr) * KVDIM + kvh * HD + kc];
    uint4 tk1 = *(const uint4*)&k[(kb + kr) * KVDIM + kvh * HD + kc + 8];
    uint4 tv0 = *(const uint4*)&vT[(kvh * HD + vd) * S_LEN + kb + vh];
    uint4 tv1 = *(const uint4*)&vT[(kvh * HD + vd) * S_LEN + kb + vh + 8];
    __syncthreads();
    *(uint4*)&Ks[0][kr * LDKV + kc]     = tk0;
    *(uint4*)&Ks[0][kr * LDKV + kc + 8] = tk1;
    *(uint4*)&Vs[0][vd * LDVT + vh]     = tv0;
    *(uint4*)&Vs[0][vd * LDVT + vh + 8] = tv1;
    __syncthreads();
    const bf16* Kc = &Ks[0][0];
    const bf16* Vc = &Vs[0][0];
#endif

    // scores: S[16x32] = Q(16x128) * K_chunk(32x128)^T  (two 16-key halves)
    v8f s0 = zero_v8f(), s1 = zero_v8f();
#pragma unroll
    for (int d = 0; d < 4; ++d) {
      v16bf b0 = load_bt_frag(&Kc[0 * LDKV + d * 32], LDKV);
      v16bf b1 = load_bt_frag(&Kc[16 * LDKV + d * 32], LDKV);
      s0 = wmma_bf16(qf[d], b0, s0);
      s1 = wmma_bf16(qf[d], b1, s1);
    }

    // scale + causal mask + chunk row-max
    float rm[8];
#pragma unroll
    for (int i = 0; i < 8; ++i) {
      const int qg = qrow0 + i + 8 * g;
      float x0 = s0[i] * scale;
      float x1 = s1[i] * scale;
      if (kb + n > qg)      x0 = -3.0e38f;
      if (kb + 16 + n > qg) x1 = -3.0e38f;
      s0[i] = x0; s1[i] = x1;
      rm[i] = fmaxf(x0, x1);
    }
#pragma unroll
    for (int mk = 1; mk < 16; mk <<= 1)
#pragma unroll
      for (int i = 0; i < 8; ++i)
        rm[i] = fmaxf(rm[i], __shfl_xor(rm[i], mk, 32));

    // online softmax update; emit P (bf16) into per-wave LDS scratch
    float fac[8], rs[8];
#pragma unroll
    for (int i = 0; i < 8; ++i) {
      const float mn = fmaxf(m_i[i], rm[i]);
      fac[i] = __expf(m_i[i] - mn);
      const float p0 = __expf(s0[i] - mn);
      const float p1 = __expf(s1[i] - mn);
      m_i[i] = mn;
      rs[i]  = p0 + p1;
      myP[(i + 8 * g) * LDP + n]      = (bf16)p0;
      myP[(i + 8 * g) * LDP + 16 + n] = (bf16)p1;
    }
#pragma unroll
    for (int mk = 1; mk < 16; mk <<= 1)
#pragma unroll
      for (int i = 0; i < 8; ++i) rs[i] += __shfl_xor(rs[i], mk, 32);
#pragma unroll
    for (int i = 0; i < 8; ++i) l_i[i] = l_i[i] * fac[i] + rs[i];
#pragma unroll
    for (int dc = 0; dc < 8; ++dc)
#pragma unroll
      for (int i = 0; i < 8; ++i) o[dc][i] *= fac[i];

    __syncthreads();  // uniform trip count; orders P write -> P read

    // O += P(16x32) * V_chunk(32keys x 128d), Vc is [d][key] -> bt fragments
    v16bf pa = load_a_frag(myP, LDP);
#pragma unroll
    for (int dc = 0; dc < 8; ++dc) {
      v16bf vb = load_bt_frag(&Vc[(dc * 16) * LDVT], LDVT);
      o[dc] = wmma_bf16(pa, vb, o[dc]);
    }
  }

  // normalize + write attn output [s, NQ*D] as bf16
  float inv[8];
#pragma unroll
  for (int i = 0; i < 8; ++i) inv[i] = 1.0f / l_i[i];
#pragma unroll
  for (int dc = 0; dc < 8; ++dc)
#pragma unroll
    for (int i = 0; i < 8; ++i) {
      attn[(qrow0 + i + 8 * g) * QDIM + h * HD + dc * 16 + n] =
          (bf16)(o[dc][i] * inv[i]);
    }
}

// ------------------------------- launcher -----------------------------------

extern "C" void kernel_launch(void* const* d_in, const int* in_sizes, int n_in,
                              void* d_out, int out_size, void* d_ws, size_t ws_size,
                              hipStream_t stream) {
  const float* hidden = (const float*)d_in[0];  // [S, H]
  const float* cos_t  = (const float*)d_in[1];  // [S, D]
  const float* sin_t  = (const float*)d_in[2];  // [S, D]
  const float* w_qkv  = (const float*)d_in[3];  // [H, FQKV]
  const float* w_o    = (const float*)d_in[4];  // [QDIM, H]
  float* out = (float*)d_out;                   // [S, H]

  char* ws = (char*)d_ws;
  bf16*  hs_b    = (bf16*)(ws);                 // [S,H]        16 MB
  bf16*  wqkvT_b = (bf16*)(ws + 16777216);      // [FQKV,H]     48 MB
  bf16*  woT_b   = (bf16*)(ws + 67108864);      // [H,QDIM]     32 MB
  float* qkv_f   = (float*)(ws + 100663296);    // [S,FQKV]     48 MB
  bf16*  q_b     = (bf16*)(ws + 150994944);     // [S,QDIM]     16 MB
  bf16*  k_b     = (bf16*)(ws + 167772160);     // [S,KVDIM]     4 MB
  bf16*  vT_b    = (bf16*)(ws + 171966464);     // [KVDIM,S]     4 MB
  bf16*  attn_b  = (bf16*)(ws + 176160768);     // [S,QDIM]     16 MB

  cast_f32_bf16<<<2048, 256, 0, stream>>>(hidden, hs_b, S_LEN * HID);
  transpose_cast<<<dim3(FQKV / 32, HID / 32), 256, 0, stream>>>(
      w_qkv, wqkvT_b, HID, FQKV);
  transpose_cast<<<dim3(HID / 32, QDIM / 32), 256, 0, stream>>>(
      w_o, woT_b, QDIM, HID);

  gemm_bf16_tn<<<dim3(FQKV / 256, S_LEN / 64), 256, 0, stream>>>(
      hs_b, wqkvT_b, qkv_f, S_LEN, FQKV, HID);

  rope_split<<<2048, 256, 0, stream>>>(qkv_f, cos_t, sin_t, q_b, k_b, vT_b);

  flash_attn<<<dim3(S_LEN / 128, NQ_H), 256, 0, stream>>>(q_b, k_b, vT_b, attn_b);

  gemm_bf16_tn<<<dim3(HID / 256, S_LEN / 64), 256, 0, stream>>>(
      attn_b, woT_b, out, S_LEN, HID, QDIM);
}
